// JambaEncoder_29068338659928
// MI455X (gfx1250) — compile-verified, matrix operand order checked by reading, not run on hardware
//
#include <hip/hip_runtime.h>
#include <math.h>

// ---------------- constants (match reference) ----------------
#define D_MODEL   128
#define HEADS     4
#define HEAD_DIM  32
#define D_INNER   256
#define STATE     16
#define KCONV     4
#define DT_RANK   8
#define BATCH     4
#define SEQ       2048
#define D_IN      256
#define LN_EPS    1e-5f

typedef __attribute__((ext_vector_type(16))) __bf16 v16bf;
typedef __attribute__((ext_vector_type(8)))  __bf16 v8bf;
typedef __attribute__((ext_vector_type(4)))  __bf16 v4bf;
typedef __attribute__((ext_vector_type(8)))  float  v8f;

// ---------------- f32 -> bf16 ----------------
__device__ inline __bf16 f2bf(float f) {
    unsigned u = __builtin_bit_cast(unsigned, f);
    unsigned r = u + 0x7FFFu + ((u >> 16) & 1u);   // round-to-nearest-even
    unsigned short hs = (unsigned short)(r >> 16);
    return __builtin_bit_cast(__bf16, hs);
}

__device__ inline void put2(v16bf& o, int i, float x, float y) {
#if __has_builtin(__builtin_amdgcn_cvt_pk_bf16_f32)
    auto p = __builtin_amdgcn_cvt_pk_bf16_f32(x, y);
    o[i] = p[0]; o[i + 1] = p[1];
#else
    o[i] = f2bf(x); o[i + 1] = f2bf(y);
#endif
}

__device__ inline v16bf pack16(float4 a0, float4 a1, float4 b0, float4 b1) {
    v16bf o;
    put2(o, 0,  a0.x, a0.y); put2(o, 2,  a0.z, a0.w);
    put2(o, 4,  a1.x, a1.y); put2(o, 6,  a1.z, a1.w);
    put2(o, 8,  b0.x, b0.y); put2(o, 10, b0.z, b0.w);
    put2(o, 12, b1.x, b1.y); put2(o, 14, b1.z, b1.w);
    return o;
}

__device__ inline v8f wmma_bf16(v16bf a, v16bf b, v8f c) {
    return __builtin_amdgcn_wmma_f32_16x16x32_bf16(
        false, a, false, b, (short)0, c, false, false);
}

// ---------------- bf16 tile loaders ----------------
// CDNA5 16-bit A/B layout: lane holds row r0+(lane&15); its 16 bf16 values are
// k = k0 + g*8 + [0..7]  and  k = k0 + 16 + g*8 + [0..7]  (g = lane>>4).
// Fast path: two 16-byte loads, no conversion, no conditionals.
__device__ inline v16bf load_nk_bf(const __bf16* __restrict__ P, int ld, int r0, int k0) {
    int lane = threadIdx.x & 31;
    int g = lane >> 4;
    const __bf16* row = P + (size_t)(r0 + (lane & 15)) * ld + k0 + g * 8;
    v8bf c0 = *(const v8bf*)(row);
    v8bf c1 = *(const v8bf*)(row + 16);
    v16bf o;
#pragma unroll
    for (int i = 0; i < 8; ++i) { o[i] = c0[i]; o[8 + i] = c1[i]; }
    return o;
}

// Guarded path: clamp addresses (always legal), zero OOB-K via AND mask.
// Loads stay unconditional -> no exec-mask branches.
__device__ inline v16bf load_nk_bf_guard(const __bf16* __restrict__ P, int ld,
                                         int r0, int k0, int Rb, int Kb) {
    int lane = threadIdx.x & 31;
    int g = lane >> 4;
    int r = r0 + (lane & 15);
    r = (r < Rb) ? r : (Rb - 1);
    const unsigned short* row = (const unsigned short*)(P + (size_t)r * ld);
    v16bf o;
#pragma unroll
    for (int j = 0; j < 8; ++j) {
        int kb = k0 + ((j & 4) << 2) + g * 8 + ((j & 3) << 1);
        int kc0 = (kb     < Kb) ? kb     : (Kb - 1);
        int kc1 = (kb + 1 < Kb) ? kb + 1 : (Kb - 1);
        unsigned short u0 = row[kc0];
        unsigned short u1 = row[kc1];
        u0 = (unsigned short)(u0 & ((kb     < Kb) ? 0xFFFFu : 0u));
        u1 = (unsigned short)(u1 & ((kb + 1 < Kb) ? 0xFFFFu : 0u));
        o[2 * j]     = __builtin_bit_cast(__bf16, u0);
        o[2 * j + 1] = __builtin_bit_cast(__bf16, u1);
    }
    return o;
}

// [K][N]-major bf16 source (V in attention), always in bounds.
__device__ inline v16bf load_kn_bf(const __bf16* __restrict__ P, int ld, int k0, int n0) {
    int lane = threadIdx.x & 31;
    int g = lane >> 4;
    int n = n0 + (lane & 15);
    v16bf o;
#pragma unroll
    for (int j = 0; j < 8; ++j) {
        int kb = k0 + ((j & 4) << 2) + g * 8 + ((j & 3) << 1);
        o[2 * j]     = P[(size_t)(kb)     * ld + n];
        o[2 * j + 1] = P[(size_t)(kb + 1) * ld + n];
    }
    return o;
}

// ---------------- f32 -> bf16 staging kernel ----------------
__launch_bounds__(256)
__global__ void cvt_kernel(const float* __restrict__ in, __bf16* __restrict__ out, int n) {
    int i = (blockIdx.x * 256 + threadIdx.x) * 4;
    if (i >= n) return;
    float4 v = *(const float4*)(in + i);
    v4bf o;
#if __has_builtin(__builtin_amdgcn_cvt_pk_bf16_f32)
    auto p0 = __builtin_amdgcn_cvt_pk_bf16_f32(v.x, v.y);
    auto p1 = __builtin_amdgcn_cvt_pk_bf16_f32(v.z, v.w);
    o[0] = p0[0]; o[1] = p0[1]; o[2] = p1[0]; o[3] = p1[1];
#else
    o[0] = f2bf(v.x); o[1] = f2bf(v.y); o[2] = f2bf(v.z); o[3] = f2bf(v.w);
#endif
    *(v4bf*)(out + i) = o;
}

// ---------------- generic GEMM: out = act(A @ W^T + bias) + resid ----------------
// A: bf16 [M][K] (lda), W: bf16 [N][K] (ldw), out/resid: f32 [M][N] (ldo)
// act: 0=none, 1=relu, 2=softplus.  GUARD=false requires M%64==0, N%64==0, K%32==0.
template <bool GUARD>
__launch_bounds__(128)
__global__ void gemm_kernel(const __bf16* __restrict__ A, const __bf16* __restrict__ W,
                            const float* __restrict__ bias, const float* __restrict__ resid,
                            float* __restrict__ out,
                            int M, int N, int K, int lda, int ldw, int ldo, int act) {
    int wave = threadIdx.x >> 5;
    int lane = threadIdx.x & 31;
    int m0 = (blockIdx.y * 4 + wave) * 16;
    int n0 = blockIdx.x * 64;
    if (m0 >= M) return;

    v8f acc[4] = {};
    for (int k0 = 0; k0 < K; k0 += 32) {
        v16bf a = GUARD ? load_nk_bf_guard(A, lda, m0, k0, M, K)
                        : load_nk_bf(A, lda, m0, k0);
#pragma unroll
        for (int t = 0; t < 4; ++t) {
            v16bf b = GUARD ? load_nk_bf_guard(W, ldw, n0 + t * 16, k0, N, K)
                            : load_nk_bf(W, ldw, n0 + t * 16, k0);
            acc[t] = wmma_bf16(a, b, acc[t]);
        }
    }

    int g = lane >> 4;
    int nl = lane & 15;
#pragma unroll
    for (int t = 0; t < 4; ++t) {
        int n = n0 + t * 16 + nl;
#pragma unroll
        for (int j = 0; j < 8; ++j) {
            int m = m0 + j + 8 * g;
            if (!GUARD || (m < M && n < N)) {
                float v = acc[t][j];
                if (bias) v += bias[n];
                if (act == 1) v = fmaxf(v, 0.f);
                else if (act == 2) v = (v > 20.f) ? v : log1pf(__expf(v));
                if (resid) v += resid[(size_t)m * ldo + n];
                out[(size_t)m * ldo + n] = v;
            }
        }
    }
}

// ---------------- attention: one wave per 16-query tile per (b,h) ----------------
// qkvb: bf16 [B*SEQ][3*D_MODEL]; obuf: f32 [B*SEQ][D_MODEL] (b,q,h,d)
__launch_bounds__(32)
__global__ void attn_kernel(const __bf16* __restrict__ qkvb, float* __restrict__ obuf) {
    extern __shared__ float S[];               // 16 * SEQ floats = 128 KB
    int qt = blockIdx.x, h = blockIdx.y, b = blockIdx.z;
    int lane = threadIdx.x;
    const __bf16* base = qkvb + (size_t)b * SEQ * (3 * D_MODEL);
    const __bf16* Q  = base + h * HEAD_DIM;
    const __bf16* Km = base + D_MODEL + h * HEAD_DIM;
    const __bf16* V  = base + 2 * D_MODEL + h * HEAD_DIM;
    int m0 = qt * 16;
    const float scale = 0.1767766953f;         // 1/sqrt(32)

    // pass 1: scores into LDS
    v16bf aq = load_nk_bf(Q, 3 * D_MODEL, m0, 0);
    for (int kt = 0; kt < SEQ / 16; ++kt) {
        v16bf bk = load_nk_bf(Km, 3 * D_MODEL, kt * 16, 0);
        v8f c = {};
        c = wmma_bf16(aq, bk, c);
        int n = kt * 16 + (lane & 15);
#pragma unroll
        for (int j = 0; j < 8; ++j) {
            int ml = j + 8 * (lane >> 4);
            S[ml * SEQ + n] = c[j] * scale;
        }
    }
    __syncthreads();

    // pass 2: softmax, float4-vectorized (row = lane&15, half-wave per half-row)
    {
        float4* S4 = (float4*)S;
        int r = lane & 15;
        int half = lane >> 4;
        int b4 = (r * SEQ + half * (SEQ / 2)) >> 2;
        float mx = -1e30f;
        for (int i = 0; i < SEQ / 8; ++i) {
            float4 v = S4[b4 + i];
            mx = fmaxf(mx, fmaxf(fmaxf(v.x, v.y), fmaxf(v.z, v.w)));
        }
        mx = fmaxf(mx, __shfl_xor(mx, 16, 32));
        float sum = 0.f;
        for (int i = 0; i < SEQ / 8; ++i) {
            float4 v = S4[b4 + i];
            v.x = __expf(v.x - mx); v.y = __expf(v.y - mx);
            v.z = __expf(v.z - mx); v.w = __expf(v.w - mx);
            S4[b4 + i] = v;
            sum += (v.x + v.y) + (v.z + v.w);
        }
        sum += __shfl_xor(sum, 16, 32);
        float inv = 1.f / sum;
        for (int i = 0; i < SEQ / 8; ++i) {
            float4 v = S4[b4 + i];
            v.x *= inv; v.y *= inv; v.z *= inv; v.w *= inv;
            S4[b4 + i] = v;
        }
    }
    __syncthreads();

    // pass 3: O = P @ V  (P from LDS as 4x b128 + pack, V gathered bf16)
    v8f o0 = {}, o1 = {};
    int g = lane >> 4;
    int mrow = lane & 15;
    for (int kt = 0; kt < SEQ / 32; ++kt) {
        int pb = mrow * SEQ + kt * 32 + g * 8;
        float4 a0 = *(const float4*)&S[pb + 0];
        float4 a1 = *(const float4*)&S[pb + 4];
        float4 b0 = *(const float4*)&S[pb + 16];
        float4 b1 = *(const float4*)&S[pb + 20];
        v16bf ap = pack16(a0, a1, b0, b1);
        v16bf vb0 = load_kn_bf(V, 3 * D_MODEL, kt * 32, 0);
        v16bf vb1 = load_kn_bf(V, 3 * D_MODEL, kt * 32, 16);
        o0 = wmma_bf16(ap, vb0, o0);
        o1 = wmma_bf16(ap, vb1, o1);
    }
    int nl = lane & 15;
#pragma unroll
    for (int j = 0; j < 8; ++j) {
        int ml = j + 8 * g;
        float* dst = obuf + ((size_t)b * SEQ + m0 + ml) * D_MODEL + h * HEAD_DIM;
        dst[nl]      = o0[j];
        dst[16 + nl] = o1[j];
    }
}

// ---------------- layernorm (row length 128), may run in place ----------------
__launch_bounds__(128)
__global__ void ln_kernel(const float* __restrict__ in, const float* __restrict__ s,
                          const float* __restrict__ bta, float* __restrict__ out) {
    __shared__ float red[128];
    int t = threadIdx.x;
    size_t row = blockIdx.x;
    float v = in[row * D_MODEL + t];
    red[t] = v; __syncthreads();
    for (int off = 64; off; off >>= 1) { if (t < off) red[t] += red[t + off]; __syncthreads(); }
    float mu = red[0] * (1.f / D_MODEL);
    __syncthreads();
    float d = v - mu;
    red[t] = d * d; __syncthreads();
    for (int off = 64; off; off >>= 1) { if (t < off) red[t] += red[t + off]; __syncthreads(); }
    float var = red[0] * (1.f / D_MODEL);
    __syncthreads();
    out[row * D_MODEL + t] = d * rsqrtf(var + LN_EPS) * s[t] + bta[t];
}

// ---------------- causal depthwise conv (K=4) + SiLU, branch-free ----------------
__global__ void conv_silu_kernel(const float* __restrict__ xz, const float* __restrict__ cw,
                                 const float* __restrict__ cb, float* __restrict__ xc) {
    int idx = blockIdx.x * blockDim.x + threadIdx.x;
    if (idx >= BATCH * SEQ * D_INNER) return;
    int c = idx & (D_INNER - 1);
    int t = (idx >> 8) & (SEQ - 1);
    int b = idx / (D_INNER * SEQ);
    const float* xcol = xz + (size_t)b * SEQ * (2 * D_INNER) + c;
    float acc = cb[c];
#pragma unroll
    for (int k = 0; k < KCONV; ++k) {
        int l = t - (KCONV - 1) + k;
        int lc = (l < 0) ? 0 : l;
        float xv = xcol[(size_t)lc * (2 * D_INNER)];     // unconditional (clamped)
        float w  = (l >= 0) ? cw[c * KCONV + k] : 0.f;   // v_cndmask
        acc = fmaf(xv, w, acc);
    }
    xc[idx] = acc / (1.f + __expf(-acc));                // silu
}

// ---------------- selective scan: one block per batch, thread = channel d ----------------
__launch_bounds__(256)
__global__ void scan_kernel(const float* __restrict__ dt, const float* __restrict__ xc,
                            const float* __restrict__ dbc, const float* __restrict__ xz,
                            const float* __restrict__ A_log, const float* __restrict__ Dp,
                            float* __restrict__ y) {
    __shared__ float bc[2 * STATE];
    int b = blockIdx.x;
    int d = threadIdx.x;
    float A[STATE];
#pragma unroll
    for (int n = 0; n < STATE; ++n) A[n] = -__expf(A_log[d * STATE + n]);
    float Dv = Dp[d];
    float hst[STATE] = {};
    for (int t = 0; t < SEQ; ++t) {
        size_t row = (size_t)b * SEQ + t;
        if (d < 2 * STATE) bc[d] = dbc[row * (DT_RANK + 2 * STATE) + DT_RANK + d];
        __syncthreads();
        float dtv = dt[row * D_INNER + d];
        float xv  = xc[row * D_INNER + d];
        float du  = dtv * xv;
        float yv  = 0.f;
#pragma unroll
        for (int n = 0; n < STATE; ++n) {
            float dA = __expf(dtv * A[n]);
            hst[n] = dA * hst[n] + du * bc[n];
            yv += hst[n] * bc[STATE + n];
        }
        yv += xv * Dv;
        float zv = xz[row * (2 * D_INNER) + D_INNER + d];
        yv *= zv / (1.f + __expf(-zv));        // * silu(z)
        y[row * D_INNER + d] = yv;
        __syncthreads();
    }
}

// ---------------- host orchestration ----------------
extern "C" void kernel_launch(void* const* d_in, const int* in_sizes, int n_in,
                              void* d_out, int out_size, void* d_ws, size_t ws_size,
                              hipStream_t stream) {
    (void)in_sizes; (void)n_in; (void)out_size; (void)ws_size;
    const float* x        = (const float*)d_in[0];
    const float* proj_w   = (const float*)d_in[1];
    const float* proj_b   = (const float*)d_in[2];
    const float* t_in_w   = (const float*)d_in[3];
    const float* t_in_b   = (const float*)d_in[4];
    const float* t_out_w  = (const float*)d_in[5];
    const float* t_out_b  = (const float*)d_in[6];
    const float* t_ff1_w  = (const float*)d_in[7];
    const float* t_ff1_b  = (const float*)d_in[8];
    const float* t_ff2_w  = (const float*)d_in[9];
    const float* t_ff2_b  = (const float*)d_in[10];
    const float* t_n1_s   = (const float*)d_in[11];
    const float* t_n1_b   = (const float*)d_in[12];
    const float* t_n2_s   = (const float*)d_in[13];
    const float* t_n2_b   = (const float*)d_in[14];
    const float* m_norm_s = (const float*)d_in[15];
    const float* m_norm_b = (const float*)d_in[16];
    const float* m_in_w   = (const float*)d_in[17];
    const float* m_conv_w = (const float*)d_in[18];
    const float* m_conv_b = (const float*)d_in[19];
    const float* m_xp_w   = (const float*)d_in[20];
    const float* m_dt_w   = (const float*)d_in[21];
    const float* m_dt_b   = (const float*)d_in[22];
    const float* m_A_log  = (const float*)d_in[23];
    const float* m_D      = (const float*)d_in[24];
    const float* m_out_w  = (const float*)d_in[25];
    const float* norm_s   = (const float*)d_in[26];
    const float* norm_b   = (const float*)d_in[27];

    const int M = BATCH * SEQ;                 // 8192 tokens
    float* ws = (float*)d_ws;
    float* h  = ws;                            // 8192*128
    float* s1 = h  + (size_t)M * D_MODEL;      // 8192*512 (qkv / xz / ff1)
    float* s2 = s1 + (size_t)M * 512;          // 8192*256 (xc / obuf)
    float* s3 = s2 + (size_t)M * D_INNER;      // 8192*40 -> reserve 64
    float* s4 = s3 + (size_t)M * 64;           // 8192*256 (dt)
    float* s5 = s4 + (size_t)M * D_INNER;      // 8192*256 (xn / y)
    __bf16* abf = (__bf16*)(s5 + (size_t)M * D_INNER);   // 8192*512 bf16 (A / qkv staging)
    __bf16* wbf = abf + (size_t)M * 512;                 // 512*512 bf16 (W staging)

    auto cvt = [&](const float* src, __bf16* dst, size_t n) {
        cvt_kernel<<<((int)(n / 4) + 255) / 256, 256, 0, stream>>>(src, dst, (int)n);
    };

    // staged GEMM: convert A slab (Mm x lda) and W (N x K) to bf16, then WMMA GEMM
    auto gemm = [&](const float* A, const float* W, const float* bias, const float* resid,
                    float* out, int Mm, int N, int K, int lda, int ldw, int ldo, int act) {
        cvt(A, abf, (size_t)Mm * lda);
        cvt(W, wbf, (size_t)N * ldw);
        dim3 grid((N + 63) / 64, (Mm + 63) / 64);
        bool exact = (Mm % 64 == 0) && (N % 64 == 0) && (K % 32 == 0);
        if (exact)
            gemm_kernel<false><<<grid, 128, 0, stream>>>(abf, wbf, bias, resid, out, Mm, N, K, lda, ldw, ldo, act);
        else
            gemm_kernel<true><<<grid, 128, 0, stream>>>(abf, wbf, bias, resid, out, Mm, N, K, lda, ldw, ldo, act);
    };

    // input projection: h = x @ proj_w^T + proj_b
    gemm(x, proj_w, proj_b, nullptr, h, M, D_MODEL, D_IN, D_IN, D_IN, D_MODEL, 0);

    // schedule for N_T=2, N_M=6: T M M M T M M M
    const char sched[8] = {'T', 'M', 'M', 'M', 'T', 'M', 'M', 'M'};
    int ti = 0, mi = 0;
    for (int li = 0; li < 8; ++li) {
        if (sched[li] == 'T') {
            gemm(h, t_in_w + (size_t)ti * 3 * D_MODEL * D_MODEL, t_in_b + ti * 3 * D_MODEL,
                 nullptr, s1, M, 3 * D_MODEL, D_MODEL, D_MODEL, D_MODEL, 3 * D_MODEL, 0);
            cvt(s1, abf, (size_t)M * 3 * D_MODEL);   // qkv -> bf16 for attention
            attn_kernel<<<dim3(SEQ / 16, HEADS, BATCH), 32, 16 * SEQ * sizeof(float), stream>>>(abf, s2);
            gemm(s2, t_out_w + (size_t)ti * D_MODEL * D_MODEL, t_out_b + ti * D_MODEL,
                 h, h, M, D_MODEL, D_MODEL, D_MODEL, D_MODEL, D_MODEL, 0);
            ln_kernel<<<M, 128, 0, stream>>>(h, t_n1_s + ti * D_MODEL, t_n1_b + ti * D_MODEL, h);
            gemm(h, t_ff1_w + (size_t)ti * 512 * D_MODEL, t_ff1_b + ti * 512,
                 nullptr, s1, M, 512, D_MODEL, D_MODEL, D_MODEL, 512, 1);
            gemm(s1, t_ff2_w + (size_t)ti * D_MODEL * 512, t_ff2_b + ti * D_MODEL,
                 h, h, M, D_MODEL, 512, 512, 512, D_MODEL, 0);
            ln_kernel<<<M, 128, 0, stream>>>(h, t_n2_s + ti * D_MODEL, t_n2_b + ti * D_MODEL, h);
            ++ti;
        } else {
            ln_kernel<<<M, 128, 0, stream>>>(h, m_norm_s + mi * D_MODEL, m_norm_b + mi * D_MODEL, s5);
            gemm(s5, m_in_w + (size_t)mi * 2 * D_INNER * D_MODEL, nullptr, nullptr,
                 s1, M, 2 * D_INNER, D_MODEL, D_MODEL, D_MODEL, 2 * D_INNER, 0);
            conv_silu_kernel<<<(M * D_INNER + 255) / 256, 256, 0, stream>>>(
                s1, m_conv_w + mi * D_INNER * KCONV, m_conv_b + mi * D_INNER, s2);
            gemm(s2, m_xp_w + (size_t)mi * (DT_RANK + 2 * STATE) * D_INNER, nullptr, nullptr,
                 s3, M, DT_RANK + 2 * STATE, D_INNER, D_INNER, D_INNER, DT_RANK + 2 * STATE, 0);
            gemm(s3, m_dt_w + (size_t)mi * D_INNER * DT_RANK, m_dt_b + mi * D_INNER, nullptr,
                 s4, M, D_INNER, DT_RANK, DT_RANK + 2 * STATE, DT_RANK, D_INNER, 2);
            scan_kernel<<<BATCH, D_INNER, 0, stream>>>(
                s4, s2, s3, s1, m_A_log + (size_t)mi * D_INNER * STATE, m_D + mi * D_INNER, s5);
            gemm(s5, m_out_w + (size_t)mi * D_MODEL * D_INNER, nullptr, h,
                 h, M, D_MODEL, D_INNER, D_INNER, D_INNER, D_MODEL, 0);
            ++mi;
        }
    }

    ln_kernel<<<M, 128, 0, stream>>>(h, norm_s, norm_b, (float*)d_out);
}
